// LLaDA2Attention_40905268527493
// MI455X (gfx1250) — matrix operand order repaired and checked by reading, not compiled, not run on hardware
//
#include <hip/hip_runtime.h>

// ---------------- problem constants (fixed by the reference) ----------------
constexpr int SEQ   = 2048;
constexpr int HIDN  = 4096;
constexpr int NH    = 32;
constexpr int NKV   = 8;
constexpr int HD    = 128;
constexpr int QKVD  = (NH + 2 * NKV) * HD;   // 6144
constexpr float RMS_EPS = 1e-6f;

// Set to 0 if the assembler rejects the async-to-LDS mnemonics.
#define USE_ASYNC_COPY 1

// ---------------- vector types for WMMA ----------------
typedef __bf16 bf16;
typedef __attribute__((ext_vector_type(16))) __bf16 v16bf;
typedef __attribute__((ext_vector_type(8)))  __bf16 v8bf;
typedef __attribute__((ext_vector_type(8)))  float  v8f;

static __device__ inline v8f v8f_zero() {
    v8f v;
#pragma unroll
    for (int i = 0; i < 8; ++i) v[i] = 0.0f;
    return v;
}

static __device__ inline v8f wmma_bf16(v16bf a, v16bf b, v8f c) {
    return __builtin_amdgcn_wmma_f32_16x16x32_bf16(
        false, a, false, b, (short)0, c, false, false);
}

static __device__ inline v16bf make_afrag(v8bf lo, v8bf hi) {
    v16bf a;
#pragma unroll
    for (int e = 0; e < 8; ++e) { a[e] = lo[e]; a[8 + e] = hi[e]; }
    return a;
}

// ---------------- async global -> LDS copy (CDNA5 ASYNCcnt path) ----------------
static __device__ inline void async_copy16(void* ldsDst, const void* gSrc) {
#if USE_ASYNC_COPY
    unsigned lds = (unsigned)(uintptr_t)ldsDst;               // low 32 bits = LDS offset
    asm volatile("global_load_async_to_lds_b128 %0, %1, off"
                 :: "v"(lds), "v"((unsigned long long)(uintptr_t)gSrc)
                 : "memory");
#else
    *(uint4*)ldsDst = *(const uint4*)gSrc;
#endif
}
static __device__ inline void async_wait() {
#if USE_ASYNC_COPY
    asm volatile("s_wait_asynccnt 0" ::: "memory");
#endif
}

// ============================================================================
// C[M,N] = A[M,K] * B[N,K]^T  (f32 memory, bf16 WMMA compute)
// 256 threads = 8 waves; block tile 128x128; wave tile 32x64.
// Double-buffered LDS; global loads register-staged ahead of compute.
// ============================================================================
__global__ __launch_bounds__(256) void gemm_bf16_wmma(
    const float* __restrict__ A, const float* __restrict__ B,
    float* __restrict__ C, int M, int N, int K)
{
    __shared__ __align__(64) bf16 As[2][128][32];   // 16 KB
    __shared__ __align__(64) bf16 Bs[2][128][32];   // 16 KB

    const int tid  = threadIdx.x;
    const int lane = tid & 31;
    const int wave = tid >> 5;
    const int g    = lane >> 4;
    const int ln   = lane & 15;

    const int bm = blockIdx.y * 128;
    const int bn = blockIdx.x * 128;
    const int wm = (wave & 3) * 32;
    const int wn = (wave >> 2) * 64;

    v8f acc[2][4];
#pragma unroll
    for (int i = 0; i < 2; ++i)
#pragma unroll
        for (int j = 0; j < 4; ++j) acc[i][j] = v8f_zero();

    float4 ra[4], rb[4];

    auto g2r = [&](int k0) {
#pragma unroll
        for (int it = 0; it < 4; ++it) {
            int e = (it * 256 + tid) * 4;          // 4 contiguous f32
            int r = e >> 5, c = e & 31;
            ra[it] = *(const float4*)&A[(size_t)(bm + r) * K + k0 + c];
            rb[it] = *(const float4*)&B[(size_t)(bn + r) * K + k0 + c];
        }
    };
    auto r2s = [&](int buf) {
#pragma unroll
        for (int it = 0; it < 4; ++it) {
            int e = (it * 256 + tid) * 4;
            int r = e >> 5, c = e & 31;
            bf16* da = &As[buf][r][c];
            da[0] = (bf16)ra[it].x; da[1] = (bf16)ra[it].y;
            da[2] = (bf16)ra[it].z; da[3] = (bf16)ra[it].w;
            bf16* db = &Bs[buf][r][c];
            db[0] = (bf16)rb[it].x; db[1] = (bf16)rb[it].y;
            db[2] = (bf16)rb[it].z; db[3] = (bf16)rb[it].w;
        }
    };

    g2r(0);
    r2s(0);
    __syncthreads();

    const int nk = K >> 5;
    for (int ki = 0; ki < nk; ++ki) {
        const int cur = ki & 1;
        if (ki + 1 < nk) g2r((ki + 1) << 5);       // next tile in flight

        v16bf af[2];
#pragma unroll
        for (int i = 0; i < 2; ++i) {
            int row = wm + i * 16 + ln;
            af[i] = *(const v16bf*)&As[cur][row][g * 16];
        }
#pragma unroll
        for (int j = 0; j < 4; ++j) {
            int col = wn + j * 16 + ln;
            v16bf bfr = *(const v16bf*)&Bs[cur][col][g * 16];
            acc[0][j] = wmma_bf16(af[0], bfr, acc[0][j]);
            acc[1][j] = wmma_bf16(af[1], bfr, acc[1][j]);
        }

        if (ki + 1 < nk) r2s(cur ^ 1);
        __syncthreads();
    }

#pragma unroll
    for (int i = 0; i < 2; ++i)
#pragma unroll
        for (int j = 0; j < 4; ++j)
#pragma unroll
            for (int r = 0; r < 8; ++r) {
                int row = bm + wm + i * 16 + g * 8 + r;
                int col = bn + wn + j * 16 + ln;
                C[(size_t)row * N + col] = acc[i][j][r];
            }
}

// ============================================================================
// RMSNorm + RoPE for q or k. One 128-thread block per (token, head) row.
// ============================================================================
__global__ __launch_bounds__(128) void qk_norm_rope(
    const float* __restrict__ qkv, const float* __restrict__ nw,
    const int* __restrict__ positions, bf16* __restrict__ out,
    int heads, int colOff)
{
    const int i   = threadIdx.x;
    const int row = blockIdx.x;
    const int t   = row / heads;
    const int hh  = row % heads;

    __shared__ float warp_s[4];
    __shared__ float xs[128];
    __shared__ float rinv_s;

    float x  = qkv[(size_t)t * QKVD + colOff + hh * HD + i];
    float ss = x * x;
#pragma unroll
    for (int off = 1; off < 32; off <<= 1) ss += __shfl_xor(ss, off, 32);
    if ((i & 31) == 0) warp_s[i >> 5] = ss;
    __syncthreads();
    if (i == 0)
        rinv_s = rsqrtf((warp_s[0] + warp_s[1] + warp_s[2] + warp_s[3]) / (float)HD + RMS_EPS);
    __syncthreads();

    float xn = x * rinv_s * nw[i];
    xs[i] = xn;
    __syncthreads();

    float y = xn;
    if (i < 64) {
        float e    = (float)(i & 31) * (1.0f / 32.0f);
        float invf = __powf(10000.0f, -e);
        float ang  = (float)positions[t] * invf;
        float c = __cosf(ang), s = __sinf(ang);
        float rot = (i < 32) ? -xs[i + 32] : xs[i - 32];
        y = xn * c + rot * s;
    }
    out[(size_t)row * HD + i] = (bf16)y;
}

// ============================================================================
// Repack V: vT[kvh][d][t] (bf16) from qkv f32
// ============================================================================
__global__ __launch_bounds__(256) void v_pack(
    const float* __restrict__ qkv, bf16* __restrict__ vT)
{
    int n   = blockIdx.x * 256 + threadIdx.x;
    int t   = n & (SEQ - 1);
    int d   = (n >> 11) & (HD - 1);
    int kvh = n >> 18;
    vT[n] = (bf16)qkv[(size_t)t * QKVD + (size_t)(NH + NKV + kvh) * HD + d];
}

// ============================================================================
// Flash attention: 1 wave = 16 query rows of one head; s-step 32.
// Double-buffered K/V tiles staged with async global->LDS copies.
// ============================================================================
__global__ __launch_bounds__(256) void attn_wmma(
    const bf16* __restrict__ q, const bf16* __restrict__ k,
    const bf16* __restrict__ vT, float* __restrict__ ctx)
{
    __shared__ __align__(64) bf16 kt[2][32][128];   // 16 KB [s][d]
    __shared__ __align__(64) bf16 vt[2][128][32];   // 16 KB [d][s]
    __shared__ __align__(64) bf16 pt[8][16][32];    //  8 KB per-wave P staging

    const int tid  = threadIdx.x;
    const int lane = tid & 31;
    const int wave = tid >> 5;
    const int g    = lane >> 4;
    const int ln   = lane & 15;

    const int h   = blockIdx.y;
    const int kvh = h >> 2;
    const int t0  = blockIdx.x * 128 + wave * 16;

    auto stage = [&](int buf, int s0) {
        int ss = tid >> 3, d0 = (tid & 7) * 16;
        async_copy16(&kt[buf][ss][d0],
                     &k[((size_t)(s0 + ss) * NKV + kvh) * HD + d0]);
        int d = tid >> 1, sc = (tid & 1) * 16;
        async_copy16(&vt[buf][d][sc],
                     &vT[((size_t)kvh * HD + d) * SEQ + s0 + sc]);
    };

    // Q fragments (16 rows x 128 d as four K=32 chunks)
    v16bf qf[4];
    {
        const bf16* qrow = q + ((size_t)(t0 + ln) * NH + h) * HD;
#pragma unroll
        for (int kk = 0; kk < 4; ++kk) {
            qf[kk] = *(const v16bf*)&qrow[kk * 32 + g * 16];
        }
    }

    float mrow[8], lrow[8];
#pragma unroll
    for (int r = 0; r < 8; ++r) { mrow[r] = -1e30f; lrow[r] = 0.0f; }
    v8f o[8];
#pragma unroll
    for (int dt = 0; dt < 8; ++dt) o[dt] = v8f_zero();

    const float scale = 0.088388347648318447f;      // 1/sqrt(128)

    stage(0, 0);
    async_wait();
    __syncthreads();

    for (int s0 = 0; s0 < SEQ; s0 += 32) {
        const int cur = (s0 >> 5) & 1;
        if (s0 + 32 < SEQ) stage(cur ^ 1, s0 + 32);   // next tile in flight

        // scores: two 16x16 S tiles over K=128
        v8f S[2];
#pragma unroll
        for (int j = 0; j < 2; ++j) {
            S[j] = v8f_zero();
#pragma unroll
            for (int kk = 0; kk < 4; ++kk) {
                v16bf bfr = *(const v16bf*)&kt[cur][j * 16 + ln][kk * 32 + g * 16];
                S[j] = wmma_bf16(qf[kk], bfr, S[j]);
            }
        }

        // online softmax; row (g*8+r) spans the 16 lanes of this half-wave
        float alpha[8];
#pragma unroll
        for (int r = 0; r < 8; ++r) {
            float v = fmaxf(S[0][r], S[1][r]);
#pragma unroll
            for (int off = 1; off < 16; off <<= 1) v = fmaxf(v, __shfl_xor(v, off, 16));
            float mnew = fmaxf(mrow[r], v * scale);
            alpha[r]   = __expf(mrow[r] - mnew);
            mrow[r]    = mnew;

            float p0 = __expf(S[0][r] * scale - mnew);
            float p1 = __expf(S[1][r] * scale - mnew);
            float rs = p0 + p1;
#pragma unroll
            for (int off = 1; off < 16; off <<= 1) rs += __shfl_xor(rs, off, 16);
            lrow[r] = lrow[r] * alpha[r] + rs;

            pt[wave][g * 8 + r][ln]      = (bf16)p0;
            pt[wave][g * 8 + r][16 + ln] = (bf16)p1;
        }

#pragma unroll
        for (int dt = 0; dt < 8; ++dt)
#pragma unroll
            for (int r = 0; r < 8; ++r) o[dt][r] *= alpha[r];

        v16bf pf;
        {
            pf = *(const v16bf*)&pt[wave][ln][g * 16];
        }
#pragma unroll
        for (int dt = 0; dt < 8; ++dt) {
            v16bf bfr = *(const v16bf*)&vt[cur][dt * 16 + ln][g * 16];
            o[dt] = wmma_bf16(pf, bfr, o[dt]);
        }

        async_wait();
        __syncthreads();
    }

#pragma unroll
    for (int r = 0; r < 8; ++r) {
        float inv = 1.0f / lrow[r];
        int   row = t0 + g * 8 + r;
#pragma unroll
        for (int dt = 0; dt < 8; ++dt)
            ctx[(size_t)row * (NH * HD) + h * HD + dt * 16 + ln] = o[dt][r] * inv;
    }
}

// ============================================================================
// Launch
// ============================================================================
extern "C" void kernel_launch(void* const* d_in, const int* in_sizes, int n_in,
                              void* d_out, int out_size, void* d_ws, size_t ws_size,
                              hipStream_t stream) {
    const int*   positions = (const int*)  d_in[0];
    const float* hs        = (const float*)d_in[1];
    const float* qkv_w     = (const float*)d_in[2];
    const float* q_norm_w  = (const float*)d_in[3];
    const float* k_norm_w  = (const float*)d_in[4];
    const float* dense_w   = (const float*)d_in[5];
    float*       out       = (float*)d_out;

    char* ws = (char*)d_ws;
    float* qkv = (float*)ws;   ws += (size_t)SEQ * QKVD * 4;        // 48 MB
    bf16*  qb  = (bf16*)ws;    ws += (size_t)SEQ * NH  * HD * 2;    // 16 MB
    bf16*  kb  = (bf16*)ws;    ws += (size_t)SEQ * NKV * HD * 2;    //  4 MB
    bf16*  vtb = (bf16*)ws;    ws += (size_t)NKV * HD * SEQ * 2;    //  4 MB
    float* ctx = (float*)ws;                                        // 32 MB

    gemm_bf16_wmma<<<dim3(QKVD / 128, SEQ / 128), 256, 0, stream>>>(
        hs, qkv_w, qkv, SEQ, QKVD, HIDN);

    qk_norm_rope<<<SEQ * NH, 128, 0, stream>>>(qkv, q_norm_w, positions, qb, NH, 0);
    qk_norm_rope<<<SEQ * NKV, 128, 0, stream>>>(qkv, k_norm_w, positions, kb, NKV, NH * HD);
    v_pack<<<(NKV * HD * SEQ) / 256, 256, 0, stream>>>(qkv, vtb);

    attn_wmma<<<dim3(SEQ / 128, NH), 256, 0, stream>>>(qb, kb, vtb, ctx);

    gemm_bf16_wmma<<<dim3(HIDN / 128, SEQ / 128), 256, 0, stream>>>(
        ctx, dense_w, out, SEQ, HIDN, HIDN);
}
